// HeteroGraphTransformerBlock_33543694582433
// MI455X (gfx1250) — compile-verified, matrix operand order checked by reading, not run on hardware
//
#include <hip/hip_runtime.h>
#include <math.h>

typedef __attribute__((ext_vector_type(16))) __bf16 v16bf;
typedef __attribute__((ext_vector_type(8)))  __bf16 v8bf;
typedef __attribute__((ext_vector_type(8)))  float  v8f;
typedef __attribute__((ext_vector_type(4)))  int    v4i;

typedef v4i __attribute__((address_space(1)))* as1_v4i_ptr;
typedef v4i __attribute__((address_space(3)))* as3_v4i_ptr;

#define SEQ 4096
#define DIM 512
#define NH  8
#define HD  64
#define FF  2048
#define NEGBIG (-1.0e30f)

#if __has_builtin(__builtin_amdgcn_global_load_async_to_lds_b128)
#define HAVE_ASYNC 1
#else
#define HAVE_ASYNC 0
#endif

// 16-byte global -> LDS copy. Async (ASYNCcnt-tracked) when the gfx1250
// builtin is available, otherwise a synchronous reg round-trip.
__device__ inline void copy16_g2l(void* dst_lds, const void* src_global) {
#if HAVE_ASYNC
  __builtin_amdgcn_global_load_async_to_lds_b128(
      (as1_v4i_ptr)(void*)src_global, (as3_v4i_ptr)dst_lds, 0, 0);
#else
  *(uint4*)dst_lds = *(const uint4*)src_global;
#endif
}

template <int N>
__device__ inline void async_wait_le() {
#if HAVE_ASYNC
#if __has_builtin(__builtin_amdgcn_s_wait_asynccnt)
  __builtin_amdgcn_s_wait_asynccnt(N);
#else
  asm volatile("s_wait_asynccnt %0" ::"i"(N) : "memory");
#endif
#endif
}

__device__ inline v8f vzero() {
  v8f z;
#pragma unroll
  for (int i = 0; i < 8; ++i) z[i] = 0.0f;
  return z;
}

__device__ inline v8f wmma_bf16(v16bf a, v16bf b, v8f c) {
  return __builtin_amdgcn_wmma_f32_16x16x32_bf16(false, a, false, b, (short)0, c,
                                                 false, false);
}

// Operand fragment from row-major global storage X[row][k] (stride ld).
__device__ inline v16bf load_frag(const __bf16* __restrict__ base, int ld,
                                  int row0, int k0, int lane) {
  int r  = row0 + (lane & 15);
  int kb = k0 + ((lane >> 4) << 3);
  const v8bf* p0 = (const v8bf*)(base + (size_t)r * ld + kb);
  const v8bf* p1 = (const v8bf*)(base + (size_t)r * ld + kb + 16);
  union { v16bf v; v8bf h[2]; } u;
  u.h[0] = *p0;
  u.h[1] = *p1;
  return u.v;
}

// Operand fragment from an LDS tile with 80-byte padded rows (32 bf16 + 16B pad):
// 16B-aligned ds_load_b128, 20-dword row stride => conflict-free over 16 lanes.
__device__ inline v16bf frag_lds(const char* buf, int row0, int lane) {
  const char* p = buf + (size_t)(row0 + (lane & 15)) * 80 + ((lane >> 4) << 4);
  union { v16bf v; v8bf h[2]; } u;
  u.h[0] = *(const v8bf*)p;
  u.h[1] = *(const v8bf*)(p + 32);
  return u.v;
}

__device__ inline v16bf pack_probs(v8f p0, v8f p1) {
  union { v16bf v; __bf16 e[16]; } u;
#pragma unroll
  for (int i = 0; i < 8; ++i) {
    u.e[i]     = (__bf16)p0[i];
    u.e[8 + i] = (__bf16)p1[i];
  }
  return u.v;
}

__device__ inline float gelu_tanh(float x) {
  float x3 = x * x * x;
  return 0.5f * x * (1.0f + tanhf(0.7978845608f * (x + 0.044715f * x3)));
}

// ---------------- weight transpose + f32->bf16 convert -------------------
__global__ void wconv_kernel(const float* __restrict__ W, __bf16* __restrict__ WT,
                             int K, int N) {
  int idx = blockIdx.x * blockDim.x + threadIdx.x;
  if (idx >= K * N) return;
  int k = idx % K;
  int n = idx / K;
  WT[idx] = (__bf16)W[(size_t)k * N + n];
}

// ---------------- layernorm (one 512-wide row per block) -----------------
__global__ void ln_kernel(const float* __restrict__ x, const int* __restrict__ mask,
                          const float* __restrict__ gamma, const float* __restrict__ beta,
                          __bf16* __restrict__ out_bf, float* __restrict__ out_f) {
  int row  = blockIdx.x;
  int tid  = threadIdx.x;
  int lane = tid & 31;
  int wave = tid >> 5;
  const float* xr = x + (size_t)row * DIM;
  float v0 = xr[tid], v1 = xr[tid + 256];
  float s  = v0 + v1;
  float ss = v0 * v0 + v1 * v1;
#pragma unroll
  for (int o = 16; o > 0; o >>= 1) {
    s  += __shfl_down(s, o);
    ss += __shfl_down(ss, o);
  }
  __shared__ float red[16];
  if (lane == 0) { red[wave] = s; red[8 + wave] = ss; }
  __syncthreads();
  float tot = 0.f, tots = 0.f;
#pragma unroll
  for (int i = 0; i < 8; ++i) { tot += red[i]; tots += red[8 + i]; }
  float mu   = tot * (1.0f / DIM);
  float var  = tots * (1.0f / DIM) - mu * mu;
  float rstd = rsqrtf(var + 1e-5f);
  float msk  = (mask[row] == 1) ? 1.0f : 0.0f;

  float y0 = ((v0 - mu) * rstd * gamma[tid]       + beta[tid])       * msk;
  float y1 = ((v1 - mu) * rstd * gamma[tid + 256] + beta[tid + 256]) * msk;
  out_bf[(size_t)row * DIM + tid]       = (__bf16)y0;
  out_bf[(size_t)row * DIM + tid + 256] = (__bf16)y1;
  if (out_f) {
    out_f[(size_t)row * DIM + tid]       = y0;
    out_f[(size_t)row * DIM + tid + 256] = y1;
  }
}

// ---------------- WMMA GEMM: async-LDS staged, double-buffered ------------
// C[M][N] = A[M][K] * BT[N][K]^T (bf16 in, f32 acc). Block tile 64x128,
// 8 waves of 32x32. A/B block tiles staged to LDS via async copies; fragments
// come from LDS (A reused 4x, B reused 2x within the block).
enum { EP_BF16_BIAS = 0, EP_BF16_BIAS_T = 1, EP_F32_BIAS_RES_MASK = 2,
       EP_BF16_GELU = 3, EP_F32_RES = 4 };

template <int MODE>
__global__ __launch_bounds__(256) void gemm_bf16_kernel(
    const __bf16* __restrict__ A, const __bf16* __restrict__ BT,
    int M, int N, int K,
    const float* __restrict__ bias, const float* __restrict__ res,
    const int* __restrict__ rmask, float scale,
    __bf16* __restrict__ out_bf, float* __restrict__ out_f) {
  // [0,5120) A buf0 | [5120,10240) A buf1 | [10240,20480) B buf0 | [20480,30720) B buf1
  __shared__ __attribute__((aligned(16))) char smem[30720];

  int tid  = threadIdx.x;
  int lane = tid & 31;
  int wave = tid >> 5;
  int mblk = blockIdx.x * 64, nblk = blockIdx.y * 128;
  int mw = (wave & 1) * 32, nw = (wave >> 1) * 32;

  auto stage = [&](int b, int k0) {
    char* Ab = smem + b * 5120;
    char* Bb = smem + 10240 + b * 10240;
    int arow = tid >> 2, ach = tid & 3;          // A: 64 rows x 64B, 1 chunk/thread
    copy16_g2l(Ab + (size_t)arow * 80 + ach * 16,
               A + (size_t)(mblk + arow) * K + k0 + ach * 8);
#pragma unroll
    for (int i = 0; i < 2; ++i) {                // B: 128 rows x 64B, 2 chunks/thread
      int idx  = tid + i * 256;
      int brow = idx >> 2, bch = idx & 3;
      copy16_g2l(Bb + (size_t)brow * 80 + bch * 16,
                 BT + (size_t)(nblk + brow) * K + k0 + bch * 8);
    }
  };

  v8f acc00 = vzero(), acc01 = vzero(), acc10 = vzero(), acc11 = vzero();
  int nsteps = K >> 5;
  stage(0, 0);
  for (int s = 0; s < nsteps; ++s) {
    int cur = s & 1;
    if (s + 1 < nsteps) {
      stage(cur ^ 1, (s + 1) << 5);   // prefetch next stage (3 async/thread)
      async_wait_le<3>();             // current stage (older 3) complete
    } else {
      async_wait_le<0>();
    }
    __syncthreads();
    const char* Ac = smem + cur * 5120;
    const char* Bc = smem + 10240 + cur * 10240;
    v16bf a0 = frag_lds(Ac, mw,      lane);
    v16bf a1 = frag_lds(Ac, mw + 16, lane);
    v16bf b0 = frag_lds(Bc, nw,      lane);
    v16bf b1 = frag_lds(Bc, nw + 16, lane);
    acc00 = wmma_bf16(a0, b0, acc00);
    acc01 = wmma_bf16(a0, b1, acc01);
    acc10 = wmma_bf16(a1, b0, acc10);
    acc11 = wmma_bf16(a1, b1, acc11);
    __syncthreads();
  }

  int m0 = mblk + mw, n0 = nblk + nw;
  v8f accs[2][2] = {{acc00, acc01}, {acc10, acc11}};
#pragma unroll
  for (int ti = 0; ti < 2; ++ti) {
#pragma unroll
    for (int tj = 0; tj < 2; ++tj) {
      int mbase = m0 + ti * 16 + ((lane >> 4) << 3);
      int n     = n0 + tj * 16 + (lane & 15);
      float b = (MODE == EP_BF16_GELU || MODE == EP_F32_RES) ? 0.0f : bias[n];
#pragma unroll
      for (int r = 0; r < 8; ++r) {
        int m   = mbase + r;
        float v = accs[ti][tj][r];
        if (MODE == EP_BF16_BIAS) {
          out_bf[(size_t)m * N + n] = (__bf16)((v + b) * scale);
        } else if (MODE == EP_BF16_BIAS_T) {
          out_bf[(size_t)n * M + m] = (__bf16)(v + b);   // store C^T (for V^T)
        } else if (MODE == EP_F32_BIAS_RES_MASK) {
          float a = (rmask[m] == 1) ? (v + b) : 0.0f;
          out_f[(size_t)m * N + n] = res[(size_t)m * N + n] + a;
        } else if (MODE == EP_BF16_GELU) {
          out_bf[(size_t)m * N + n] = (__bf16)gelu_tanh(v);
        } else {
          out_f[(size_t)m * N + n] = res[(size_t)m * N + n] + v;
        }
      }
    }
  }
}

// ---------------- masked flash attention ---------------------------------
// Block = one 16-row target tile; wave = one head. Logits are computed
// transposed (L^T = K·Q^T) so the softmaxed fragment is directly the B
// operand of ctx^T = V^T·P^T (no cross-lane transpose). The head-invariant
// edge-mask tile is async-staged to LDS once per block (8x traffic cut),
// double-buffered across s-chunks.
__global__ __launch_bounds__(256) void attn_kernel(
    const __bf16* __restrict__ q,    // [T][512] (pre-scaled 1/8)
    const __bf16* __restrict__ kmat, // [S][512]
    const __bf16* __restrict__ vT,   // [512][S]
    const int* __restrict__ edge_mask, // [T][S]
    __bf16* __restrict__ ctx) {      // [T][512]
  __shared__ __attribute__((aligned(16))) int msk[2][16 * 36]; // 144B rows

  int tid  = threadIdx.x;
  int lane = tid & 31;
  int head = tid >> 5;               // 8 waves = 8 heads
  int t0   = blockIdx.x * 16;
  int hcol = head * HD;

  auto stage_mask = [&](int b, int s0) {
    if (tid < 128) {                 // 16 rows x 128B, 1 chunk/thread
      int row = tid >> 3, ch = tid & 7;
      copy16_g2l((char*)&msk[b][0] + (size_t)row * 144 + ch * 16,
                 edge_mask + (size_t)(t0 + row) * SEQ + s0 + ch * 4);
    }
  };

  v16bf qf0 = load_frag(q, DIM, t0, hcol,      lane);
  v16bf qf1 = load_frag(q, DIM, t0, hcol + 32, lane);

  v8f acc[4];
#pragma unroll
  for (int i = 0; i < 4; ++i) acc[i] = vzero();

  float m_run = -3.0e38f, l_run = 0.0f;
  int tl       = lane & 15;
  int srow_off = (lane >> 4) << 3;

  stage_mask(0, 0);
  for (int s0 = 0; s0 < SEQ; s0 += 32) {
    int cur = (s0 >> 5) & 1;
    if (s0 + 32 < SEQ) {
      stage_mask(cur ^ 1, s0 + 32);
      async_wait_le<1>();
      __builtin_prefetch(kmat + (size_t)(s0 + 32 + tl) * DIM + hcol, 0, 1);
    } else {
      async_wait_le<0>();
    }
    __syncthreads();
    const int* mrowbase = &msk[cur][0] + tl * 36 + srow_off;

    v8f lg[2];
#pragma unroll
    for (int st = 0; st < 2; ++st) {
      v16bf kf0 = load_frag(kmat, DIM, s0 + st * 16, hcol,      lane);
      v16bf kf1 = load_frag(kmat, DIM, s0 + st * 16, hcol + 32, lane);
      v8f c = vzero();
      c = wmma_bf16(kf0, qf0, c);
      c = wmma_bf16(kf1, qf1, c);
      const int* mrow = mrowbase + st * 16;
#pragma unroll
      for (int r = 0; r < 8; ++r) c[r] = (mrow[r] != 0) ? c[r] : NEGBIG;
      lg[st] = c;
    }
    // online softmax over s: within-lane + one cross-half shuffle
    float cmax = -3.0e38f;
#pragma unroll
    for (int r = 0; r < 8; ++r) cmax = fmaxf(cmax, fmaxf(lg[0][r], lg[1][r]));
    cmax = fmaxf(cmax, __shfl_xor(cmax, 16));
    float m_new = fmaxf(m_run, cmax);
    float alpha = __expf(m_run - m_new);
    float lsum = 0.0f;
#pragma unroll
    for (int st = 0; st < 2; ++st)
#pragma unroll
      for (int r = 0; r < 8; ++r) {
        float p = __expf(lg[st][r] - m_new);
        lg[st][r] = p;
        lsum += p;
      }
    lsum += __shfl_xor(lsum, 16);
    l_run = l_run * alpha + lsum;
    m_run = m_new;
#pragma unroll
    for (int i = 0; i < 4; ++i) acc[i] *= alpha;

    v16bf pf = pack_probs(lg[0], lg[1]);
#pragma unroll
    for (int dt = 0; dt < 4; ++dt) {
      v16bf vf = load_frag(vT, SEQ, hcol + dt * 16, s0, lane);
      acc[dt] = wmma_bf16(vf, pf, acc[dt]);
    }
    __syncthreads();
  }

  float inv = 1.0f / l_run;
#pragma unroll
  for (int dt = 0; dt < 4; ++dt)
#pragma unroll
    for (int r = 0; r < 8; ++r) {
      int d = hcol + dt * 16 + srow_off + r;
      ctx[(size_t)(t0 + tl) * DIM + d] = (__bf16)(acc[dt][r] * inv);
    }
}

// -------------------------------------------------------------------------
extern "C" void kernel_launch(void* const* d_in, const int* in_sizes, int n_in,
                              void* d_out, int out_size, void* d_ws, size_t ws_size,
                              hipStream_t stream) {
  const float* h_source    = (const float*)d_in[0];
  const float* h_target    = (const float*)d_in[1];
  const int*   edge_mask   = (const int*)d_in[2];
  const int*   source_mask = (const int*)d_in[3];
  const int*   target_mask = (const int*)d_in[4];
  const float* ln1_scale   = (const float*)d_in[5];
  const float* ln1_bias    = (const float*)d_in[6];
  const float* Wq = (const float*)d_in[7];  const float* bq = (const float*)d_in[8];
  const float* Wk = (const float*)d_in[9];  const float* bk = (const float*)d_in[10];
  const float* Wv = (const float*)d_in[11]; const float* bv = (const float*)d_in[12];
  const float* Wo = (const float*)d_in[13]; const float* bo = (const float*)d_in[14];
  const float* ln2_scale = (const float*)d_in[15];
  const float* ln2_bias  = (const float*)d_in[16];
  const float* W1 = (const float*)d_in[17];
  const float* W2 = (const float*)d_in[18];
  float* out = (float*)d_out;

  char* ws = (char*)d_ws;
  size_t off = 0;
  auto alloc = [&](size_t bytes) {
    void* p = ws + off;
    off = (off + bytes + 255) & ~(size_t)255;
    return p;
  };
  const size_t SD = (size_t)SEQ * DIM;
  __bf16* hs_bf  = (__bf16*)alloc(SD * 2);
  __bf16* ht_bf  = (__bf16*)alloc(SD * 2);
  float*  ht_ln  = (float*)alloc(SD * 4);
  __bf16* q_bf   = (__bf16*)alloc(SD * 2);
  __bf16* k_bf   = (__bf16*)alloc(SD * 2);
  __bf16* vT_bf  = (__bf16*)alloc(SD * 2);
  __bf16* ctx_bf = (__bf16*)alloc(SD * 2);
  float*  y_f    = (float*)alloc(SD * 4);
  __bf16* h2_bf  = (__bf16*)alloc(SD * 2);
  float*  h2_f   = (float*)alloc(SD * 4);
  __bf16* ffn_bf = (__bf16*)alloc((size_t)SEQ * FF * 2);
  __bf16* WqT = (__bf16*)alloc((size_t)DIM * DIM * 2);
  __bf16* WkT = (__bf16*)alloc((size_t)DIM * DIM * 2);
  __bf16* WvT = (__bf16*)alloc((size_t)DIM * DIM * 2);
  __bf16* WoT = (__bf16*)alloc((size_t)DIM * DIM * 2);
  __bf16* W1T = (__bf16*)alloc((size_t)DIM * FF * 2);
  __bf16* W2T = (__bf16*)alloc((size_t)FF * DIM * 2);

  int wgrid = (DIM * DIM + 255) / 256;
  wconv_kernel<<<wgrid, 256, 0, stream>>>(Wq, WqT, DIM, DIM);
  wconv_kernel<<<wgrid, 256, 0, stream>>>(Wk, WkT, DIM, DIM);
  wconv_kernel<<<wgrid, 256, 0, stream>>>(Wv, WvT, DIM, DIM);
  wconv_kernel<<<wgrid, 256, 0, stream>>>(Wo, WoT, DIM, DIM);
  int fgrid = (DIM * FF + 255) / 256;
  wconv_kernel<<<fgrid, 256, 0, stream>>>(W1, W1T, DIM, FF);
  wconv_kernel<<<fgrid, 256, 0, stream>>>(W2, W2T, FF, DIM);

  ln_kernel<<<SEQ, 256, 0, stream>>>(h_source, source_mask, ln1_scale, ln1_bias,
                                     hs_bf, nullptr);
  ln_kernel<<<SEQ, 256, 0, stream>>>(h_target, target_mask, ln1_scale, ln1_bias,
                                     ht_bf, ht_ln);

  dim3 gproj(SEQ / 64, DIM / 128);
  gemm_bf16_kernel<EP_BF16_BIAS><<<gproj, 256, 0, stream>>>(
      ht_bf, WqT, SEQ, DIM, DIM, bq, nullptr, nullptr, 0.125f, q_bf, nullptr);
  gemm_bf16_kernel<EP_BF16_BIAS><<<gproj, 256, 0, stream>>>(
      hs_bf, WkT, SEQ, DIM, DIM, bk, nullptr, nullptr, 1.0f, k_bf, nullptr);
  gemm_bf16_kernel<EP_BF16_BIAS_T><<<gproj, 256, 0, stream>>>(
      hs_bf, WvT, SEQ, DIM, DIM, bv, nullptr, nullptr, 1.0f, vT_bf, nullptr);

  attn_kernel<<<SEQ / 16, 256, 0, stream>>>(q_bf, k_bf, vT_bf, edge_mask, ctx_bf);

  gemm_bf16_kernel<EP_F32_BIAS_RES_MASK><<<gproj, 256, 0, stream>>>(
      ctx_bf, WoT, SEQ, DIM, DIM, bo, ht_ln, target_mask, 1.0f, nullptr, y_f);

  ln_kernel<<<SEQ, 256, 0, stream>>>(y_f, target_mask, ln2_scale, ln2_bias,
                                     h2_bf, h2_f);

  dim3 gff1(SEQ / 64, FF / 128);
  gemm_bf16_kernel<EP_BF16_GELU><<<gff1, 256, 0, stream>>>(
      h2_bf, W1T, SEQ, FF, DIM, nullptr, nullptr, nullptr, 1.0f, ffn_bf, nullptr);

  gemm_bf16_kernel<EP_F32_RES><<<gproj, 256, 0, stream>>>(
      ffn_bf, W2T, SEQ, DIM, FF, nullptr, h2_f, nullptr, 1.0f, nullptr, out);
}
